// EnhancedGatedFusion_4715874091318
// MI455X (gfx1250) — matrix operand order repaired and checked by reading, not compiled, not run on hardware
//
#include <hip/hip_runtime.h>
#include <hip/hip_bf16.h>
#include <stdint.h>

#define T_TOK 8192
#define DDIM  2048
#define NEXP  8

typedef __attribute__((ext_vector_type(16))) __bf16         v16bf;
typedef __attribute__((ext_vector_type(16))) unsigned short v16u;
typedef __attribute__((ext_vector_type(8)))  unsigned short u16x8;
typedef __attribute__((ext_vector_type(8)))  float          v8f;
typedef __attribute__((ext_vector_type(4)))  unsigned int   v4ui;
typedef __attribute__((ext_vector_type(8)))  int            v8si;
typedef __attribute__((ext_vector_type(4)))  int            v4si;

#if defined(__has_builtin)
#  if __has_builtin(__builtin_amdgcn_tensor_load_to_lds) && \
      __has_builtin(__builtin_amdgcn_s_wait_tensorcnt)
#    define HAVE_TDM 1
#  endif
#endif
#ifndef HAVE_TDM
#  define HAVE_TDM 0
#endif

__device__ __forceinline__ unsigned short f32_bf16(float f) {
    unsigned int u = __float_as_uint(f);
    u += 0x7FFFu + ((u >> 16) & 1u);            // round-to-nearest-even
    return (unsigned short)(u >> 16);
}

#if HAVE_TDM
// TDM 2D tile load: 128 rows x 32 bf16 cols from a [rows x 2048] bf16 tensor
// (row stride 2048 elems) into LDS laid out [row][col] contiguously.
__device__ __forceinline__ void tdm_load_tile_128x32(const unsigned short* gsrc,
                                                     unsigned short* ldst,
                                                     unsigned int rows) {
    unsigned long long ga = (unsigned long long)(uintptr_t)gsrc;
    unsigned int ldsa = (unsigned int)(uintptr_t)ldst;   // flat LDS: low 32 = LDS byte addr
    v4ui g0 = { 1u,                                      // count = 1 valid descriptor
                ldsa,                                    // lds_addr
                (unsigned int)ga,                        // global_addr[31:0]
                ((unsigned int)(ga >> 32) & 0x01FFFFFFu) | (2u << 30) };  // addr[56:32] | type=2
    v8si g1 = { 0x00010000,                // workgroup_mask=0, data_size=1 (2 bytes)
                (int)(2048u << 16),        // tensor_dim0 = 2048 (lo16 in bits[31:16])
                (int)(rows << 16),         // tensor_dim0 hi=0 | tensor_dim1 lo16 = rows
                (int)(32u << 16),          // tensor_dim1 hi=0 | tile_dim0 = 32
                128,                       // tile_dim1 = 128, tile_dim2 = 0
                2048,                      // tensor_dim0_stride[31:0] = 2048 elems
                0, 0 };
    v4si z4 = { 0, 0, 0, 0 };
#if __has_include(<hip/amd_detail/amd_gfx1250_TDM.h>)
    v8si z8 = { 0, 0, 0, 0, 0, 0, 0, 0 };
    __builtin_amdgcn_tensor_load_to_lds(g0, g1, z4, z4, z8, 0);
#else
    __builtin_amdgcn_tensor_load_to_lds(g0, g1, z4, z4, 0);
#endif
}
#endif

// ---------------------------------------------------------------- conversions
__global__ void cvt_bf16_kernel(const float* __restrict__ in,
                                unsigned short* __restrict__ out, int n) {
    int i = blockIdx.x * blockDim.x + threadIdx.x;
    if (i < n) out[i] = f32_bf16(in[i]);
}

// f32 [z][k][n] (k-major)  ->  bf16 [z][n][k] (n-major, transposed) via LDS tile
__global__ __launch_bounds__(256) void cvt_transpose_kernel(
    const float* __restrict__ in, unsigned short* __restrict__ out) {
    __shared__ float t[32][33];
    const float* src = in + (size_t)blockIdx.z * DDIM * DDIM;
    unsigned short* dst = out + (size_t)blockIdx.z * DDIM * DDIM;
    const int tx = threadIdx.x;              // 0..31
    const int ty = threadIdx.y;              // 0..7
    const int k0 = blockIdx.y * 32, n0 = blockIdx.x * 32;
#pragma unroll
    for (int j = 0; j < 32; j += 8)
        t[ty + j][tx] = src[(size_t)(k0 + ty + j) * DDIM + n0 + tx];
    __syncthreads();
#pragma unroll
    for (int j = 0; j < 32; j += 8)
        dst[(size_t)(n0 + ty + j) * DDIM + k0 + tx] = f32_bf16(t[tx][ty + j]);
}

// ---------------------------------------------------------------- router
__global__ __launch_bounds__(256) void router_kernel(
    const float* __restrict__ x, const float* __restrict__ rw,
    const float* __restrict__ rb, float* __restrict__ fullw) {
    const int lane = threadIdx.x & 31;
    const int wave = threadIdx.x >> 5;
    const int t = blockIdx.x * 8 + wave;
    if (t >= T_TOK) return;

    float acc[NEXP];
#pragma unroll
    for (int e = 0; e < NEXP; ++e) acc[e] = 0.f;

    const float* xr = x + (size_t)t * DDIM;
    for (int d = lane; d < DDIM; d += 32) {
        float xv = xr[d];
        const float* r = rw + (size_t)d * NEXP;
#pragma unroll
        for (int e = 0; e < NEXP; ++e) acc[e] = fmaf(xv, r[e], acc[e]);
    }
#pragma unroll
    for (int e = 0; e < NEXP; ++e) {
        for (int m = 16; m >= 1; m >>= 1) acc[e] += __shfl_xor(acc[e], m, 32);
    }
    if (lane == 0) {
        float lg[NEXP];
#pragma unroll
        for (int e = 0; e < NEXP; ++e) lg[e] = acc[e] + rb[e];
        int i1 = 0;
        for (int e = 1; e < NEXP; ++e) if (lg[e] > lg[i1]) i1 = e;
        int i2 = (i1 == 0) ? 1 : 0;
        for (int e = 0; e < NEXP; ++e)
            if (e != i1 && lg[e] > lg[i2]) i2 = e;
        float ex = __expf(lg[i2] - lg[i1]);          // softmax over top-2
        float w1 = 1.0f / (1.0f + ex);
        float w2 = ex * w1;
        float* o = fullw + (size_t)t * NEXP;
#pragma unroll
        for (int e = 0; e < NEXP; ++e) o[e] = 0.f;
        o[i1] = w1;
        o[i2] = w2;
    }
}

// ---------------------------------------------------------------- fused MoE GEMM
// comb[t,f] = sum_e fullw[t,e] * silu( x[t,:] @ W_e[:,f] + eb[e,f] )
// B (weights) pre-transposed: wtbf[e][n][k]
__global__ __launch_bounds__(256) void moe_expert_kernel(
    const unsigned short* __restrict__ xbf,   // [T,D] bf16
    const unsigned short* __restrict__ wtbf,  // [E,D,D] bf16, [n][k]
    const float* __restrict__ ebias,          // [E,D]
    const float* __restrict__ fullw,          // [T,E]
    unsigned short* __restrict__ comb) {      // [T,D] bf16
    __shared__ __align__(16) unsigned short As[2][128 * 32];   // [m][k]
    __shared__ __align__(16) unsigned short Bs[2][128 * 32];   // [n][k]
    __shared__ float wS[128 * NEXP];

    const int tid  = threadIdx.x;
    const int lane = tid & 31;
    const int wave = tid >> 5;
    const int half = lane >> 4;
    const int ln16 = lane & 15;

    const int rowBlk = blockIdx.x * 128;     // token rows
    const int colBlk = blockIdx.y * 128;     // feature cols
    const int waveM  = (wave & 3) * 32;
    const int waveN  = (wave >> 2) * 64;

    for (int i = tid; i < 128 * NEXP; i += 256)
        wS[i] = fullw[(size_t)rowBlk * NEXP + i];

    v8f comb_acc[2][4];
#pragma unroll
    for (int mi = 0; mi < 2; ++mi)
#pragma unroll
        for (int ni = 0; ni < 4; ++ni)
#pragma unroll
            for (int i = 0; i < 8; ++i) comb_acc[mi][ni][i] = 0.f;

    const int STEPS = NEXP * (DDIM / 32);

    auto stage = [&](int s, int buf) {
        const int e  = s >> 6;
        const int kt = (s & 63) << 5;
#if HAVE_TDM
        if (tid < 32)        // wave 0: A tile via Tensor Data Mover
            tdm_load_tile_128x32(xbf + (size_t)rowBlk * DDIM + kt, &As[buf][0], 8192u);
        else if (tid < 64)   // wave 1: B tile via Tensor Data Mover
            tdm_load_tile_128x32(wtbf + (size_t)e * DDIM * DDIM +
                                 (size_t)colBlk * DDIM + kt, &Bs[buf][0], 2048u);
#else
        int r = tid >> 1, c = (tid & 1) << 4;
        { const u16x8* sp = (const u16x8*)(xbf + (size_t)(rowBlk + r) * DDIM + kt + c);
          u16x8* dp = (u16x8*)(&As[buf][r * 32 + c]); dp[0] = sp[0]; dp[1] = sp[1]; }
        { const u16x8* sp = (const u16x8*)(wtbf + (size_t)e * DDIM * DDIM +
                                           (size_t)(colBlk + r) * DDIM + kt + c);
          u16x8* dp = (u16x8*)(&Bs[buf][r * 32 + c]); dp[0] = sp[0]; dp[1] = sp[1]; }
#endif
    };

    v8f acc[2][4];
    int cur = 0;
    stage(0, 0);                                     // pipeline prologue
    for (int s = 0; s < STEPS; ++s) {
        const int e = s >> 6;
        __syncthreads();                             // all reads of buf[cur^1] done
        if (s + 1 < STEPS) {
            stage(s + 1, cur ^ 1);                   // DMA next tile, overlaps WMMA below
            if (s + 2 < STEPS && tid >= 64 && tid < 192) {   // warm GL2 two steps ahead
                const int s2 = s + 2, e2 = s2 >> 6, kt2 = (s2 & 63) << 5;
                __builtin_prefetch(wtbf + (size_t)e2 * DDIM * DDIM +
                                   (size_t)(colBlk + tid - 64) * DDIM + kt2, 0, 0);
            }
#if HAVE_TDM
            if (tid < 64) __builtin_amdgcn_s_wait_tensorcnt(1);  // current buf retired
#endif
        } else {
#if HAVE_TDM
            if (tid < 64) __builtin_amdgcn_s_wait_tensorcnt(0);
#endif
        }
        __syncthreads();                             // buf[cur] visible to all waves

        if ((s & 63) == 0) {
#pragma unroll
            for (int mi = 0; mi < 2; ++mi)
#pragma unroll
                for (int ni = 0; ni < 4; ++ni)
#pragma unroll
                    for (int i = 0; i < 8; ++i) acc[mi][ni][i] = 0.f;
        }

        v16u a[2], b[4];
#pragma unroll
        for (int mi = 0; mi < 2; ++mi) {   // A frag: K = 16*half + j
            const unsigned short* p = &As[cur][(waveM + mi * 16 + ln16) * 32 + half * 16];
            u16x8 lo = *(const u16x8*)p;
            u16x8 hi = *(const u16x8*)(p + 8);
#pragma unroll
            for (int j = 0; j < 8; ++j) { a[mi][j] = lo[j]; a[mi][8 + j] = hi[j]; }
        }
#pragma unroll
        for (int ni = 0; ni < 4; ++ni) {   // B frag: K = 16*half + j
            const unsigned short* p = &Bs[cur][(waveN + ni * 16 + ln16) * 32 + half * 16];
            u16x8 lo = *(const u16x8*)p;
            u16x8 hi = *(const u16x8*)(p + 8);
#pragma unroll
            for (int j = 0; j < 8; ++j) { b[ni][j] = lo[j]; b[ni][8 + j] = hi[j]; }
        }
#pragma unroll
        for (int mi = 0; mi < 2; ++mi)
#pragma unroll
            for (int ni = 0; ni < 4; ++ni)
                acc[mi][ni] = __builtin_amdgcn_wmma_f32_16x16x32_bf16(
                    false, __builtin_bit_cast(v16bf, a[mi]),
                    false, __builtin_bit_cast(v16bf, b[ni]),
                    (short)0, acc[mi][ni], false, false);

        if ((s & 63) == 63) {   // per-expert epilogue: bias + SiLU + gate accumulate
#pragma unroll
            for (int ni = 0; ni < 4; ++ni) {
                float bias = ebias[(size_t)e * DDIM + colBlk + waveN + ni * 16 + ln16];
#pragma unroll
                for (int mi = 0; mi < 2; ++mi) {
#pragma unroll
                    for (int i = 0; i < 8; ++i) {
                        float h = acc[mi][ni][i] + bias;
                        float sg = h * __builtin_amdgcn_rcpf(1.0f + __expf(-h));
                        float w = wS[(waveM + mi * 16 + half * 8 + i) * NEXP + e];
                        comb_acc[mi][ni][i] += w * sg;
                    }
                }
            }
        }
        cur ^= 1;
    }

    // write combined expert output (bf16 for the projection GEMM)
#pragma unroll
    for (int mi = 0; mi < 2; ++mi)
#pragma unroll
        for (int ni = 0; ni < 4; ++ni)
#pragma unroll
            for (int i = 0; i < 8; ++i) {
                int row = rowBlk + waveM + mi * 16 + half * 8 + i;
                int col = colBlk + waveN + ni * 16 + ln16;
                comb[(size_t)row * DDIM + col] = f32_bf16(comb_acc[mi][ni][i]);
            }
}

// ---------------------------------------------------------------- projection GEMM
// y = comb @ proj_w + proj_b + x   (residual fused); proj_w pre-transposed [n][k]
__global__ __launch_bounds__(256) void proj_gemm_kernel(
    const unsigned short* __restrict__ abf,   // [T,D] bf16
    const unsigned short* __restrict__ wtbf,  // [D,D] bf16, [n][k]
    const float* __restrict__ bias,           // [D]
    const float* __restrict__ xres,           // [T,D] f32
    float* __restrict__ y) {                  // [T,D] f32
    __shared__ __align__(16) unsigned short As[2][128 * 32];
    __shared__ __align__(16) unsigned short Bs[2][128 * 32];

    const int tid  = threadIdx.x;
    const int lane = tid & 31;
    const int wave = tid >> 5;
    const int half = lane >> 4;
    const int ln16 = lane & 15;

    const int rowBlk = blockIdx.x * 128;
    const int colBlk = blockIdx.y * 128;
    const int waveM  = (wave & 3) * 32;
    const int waveN  = (wave >> 2) * 64;

    v8f acc[2][4];
#pragma unroll
    for (int mi = 0; mi < 2; ++mi)
#pragma unroll
        for (int ni = 0; ni < 4; ++ni)
#pragma unroll
            for (int i = 0; i < 8; ++i) acc[mi][ni][i] = 0.f;

    const int STEPS = DDIM / 32;

    auto stage = [&](int s, int buf) {
        const int kt = s << 5;
#if HAVE_TDM
        if (tid < 32)
            tdm_load_tile_128x32(abf + (size_t)rowBlk * DDIM + kt, &As[buf][0], 8192u);
        else if (tid < 64)
            tdm_load_tile_128x32(wtbf + (size_t)colBlk * DDIM + kt, &Bs[buf][0], 2048u);
#else
        int r = tid >> 1, c = (tid & 1) << 4;
        { const u16x8* sp = (const u16x8*)(abf + (size_t)(rowBlk + r) * DDIM + kt + c);
          u16x8* dp = (u16x8*)(&As[buf][r * 32 + c]); dp[0] = sp[0]; dp[1] = sp[1]; }
        { const u16x8* sp = (const u16x8*)(wtbf + (size_t)(colBlk + r) * DDIM + kt + c);
          u16x8* dp = (u16x8*)(&Bs[buf][r * 32 + c]); dp[0] = sp[0]; dp[1] = sp[1]; }
#endif
    };

    int cur = 0;
    stage(0, 0);
    for (int s = 0; s < STEPS; ++s) {
        __syncthreads();
        if (s + 1 < STEPS) {
            stage(s + 1, cur ^ 1);
            if (s + 2 < STEPS && tid >= 64 && tid < 192) {
                const int kt2 = (s + 2) << 5;
                __builtin_prefetch(wtbf + (size_t)(colBlk + tid - 64) * DDIM + kt2, 0, 0);
            }
#if HAVE_TDM
            if (tid < 64) __builtin_amdgcn_s_wait_tensorcnt(1);
#endif
        } else {
#if HAVE_TDM
            if (tid < 64) __builtin_amdgcn_s_wait_tensorcnt(0);
#endif
        }
        __syncthreads();

        v16u a[2], b[4];
#pragma unroll
        for (int mi = 0; mi < 2; ++mi) {
            const unsigned short* p = &As[cur][(waveM + mi * 16 + ln16) * 32 + half * 16];
            u16x8 lo = *(const u16x8*)p;
            u16x8 hi = *(const u16x8*)(p + 8);
#pragma unroll
            for (int j = 0; j < 8; ++j) { a[mi][j] = lo[j]; a[mi][8 + j] = hi[j]; }
        }
#pragma unroll
        for (int ni = 0; ni < 4; ++ni) {
            const unsigned short* p = &Bs[cur][(waveN + ni * 16 + ln16) * 32 + half * 16];
            u16x8 lo = *(const u16x8*)p;
            u16x8 hi = *(const u16x8*)(p + 8);
#pragma unroll
            for (int j = 0; j < 8; ++j) { b[ni][j] = lo[j]; b[ni][8 + j] = hi[j]; }
        }
#pragma unroll
        for (int mi = 0; mi < 2; ++mi)
#pragma unroll
            for (int ni = 0; ni < 4; ++ni)
                acc[mi][ni] = __builtin_amdgcn_wmma_f32_16x16x32_bf16(
                    false, __builtin_bit_cast(v16bf, a[mi]),
                    false, __builtin_bit_cast(v16bf, b[ni]),
                    (short)0, acc[mi][ni], false, false);
        cur ^= 1;
    }

#pragma unroll
    for (int ni = 0; ni < 4; ++ni) {
        int col = colBlk + waveN + ni * 16 + ln16;
        float bc = bias[col];
#pragma unroll
        for (int mi = 0; mi < 2; ++mi)
#pragma unroll
            for (int i = 0; i < 8; ++i) {
                int row = rowBlk + waveM + mi * 16 + half * 8 + i;
                y[(size_t)row * DDIM + col] =
                    acc[mi][ni][i] + bc + xres[(size_t)row * DDIM + col];
            }
    }
}

// ---------------------------------------------------------------- RMSNorm
__global__ __launch_bounds__(256) void rmsnorm_kernel(
    const float* __restrict__ y, const float* __restrict__ nw,
    float* __restrict__ out) {
    __shared__ float red[256];
    const int row = blockIdx.x;
    const float* yr = y + (size_t)row * DDIM;
    float s = 0.f;
    for (int d = threadIdx.x; d < DDIM; d += 256) {
        float v = yr[d];
        s += v * v;
    }
    red[threadIdx.x] = s;
    __syncthreads();
    for (int st = 128; st > 0; st >>= 1) {
        if (threadIdx.x < st) red[threadIdx.x] += red[threadIdx.x + st];
        __syncthreads();
    }
    float rinv = __builtin_amdgcn_rsqf(red[0] * (1.0f / DDIM) + 1e-6f);
    for (int d = threadIdx.x; d < DDIM; d += 256)
        out[(size_t)row * DDIM + d] = nw[d] * yr[d] * rinv;
}

// ---------------------------------------------------------------- launch
extern "C" void kernel_launch(void* const* d_in, const int* in_sizes, int n_in,
                              void* d_out, int out_size, void* d_ws, size_t ws_size,
                              hipStream_t stream) {
    (void)in_sizes; (void)n_in; (void)out_size; (void)ws_size;
    const float* x  = (const float*)d_in[0];
    const float* rw = (const float*)d_in[1];
    const float* rb = (const float*)d_in[2];
    const float* ew = (const float*)d_in[3];
    const float* eb = (const float*)d_in[4];
    const float* pw = (const float*)d_in[5];
    const float* pb = (const float*)d_in[6];
    const float* nw = (const float*)d_in[7];
    float* out = (float*)d_out;

    char* ws = (char*)d_ws;
    size_t off = 0;
    auto take = [&](size_t bytes) {
        void* p = ws + off;
        off = (off + bytes + 255) & ~(size_t)255;
        return p;
    };
    unsigned short* xbf    = (unsigned short*)take((size_t)T_TOK * DDIM * 2);
    unsigned short* wtbf   = (unsigned short*)take((size_t)NEXP * DDIM * DDIM * 2);
    unsigned short* pwt    = (unsigned short*)take((size_t)DDIM * DDIM * 2);
    float*          fullw  = (float*)take((size_t)T_TOK * NEXP * 4);
    unsigned short* combbf = (unsigned short*)take((size_t)T_TOK * DDIM * 2);
    float*          yb     = (float*)take((size_t)T_TOK * DDIM * 4);

    int n1 = T_TOK * DDIM;
    cvt_bf16_kernel<<<(n1 + 255) / 256, 256, 0, stream>>>(x, xbf, n1);

    dim3 tblk(32, 8);
    dim3 tgrdE(DDIM / 32, DDIM / 32, NEXP);
    cvt_transpose_kernel<<<tgrdE, tblk, 0, stream>>>(ew, wtbf);   // [e][k][n] -> [e][n][k]
    dim3 tgrdP(DDIM / 32, DDIM / 32, 1);
    cvt_transpose_kernel<<<tgrdP, tblk, 0, stream>>>(pw, pwt);    // [k][n] -> [n][k]

    router_kernel<<<T_TOK / 8, 256, 0, stream>>>(x, rw, rb, fullw);

    dim3 grid(T_TOK / 128, DDIM / 128);
    moe_expert_kernel<<<grid, 256, 0, stream>>>(xbf, wtbf, eb, fullw, combbf);
    proj_gemm_kernel<<<grid, 256, 0, stream>>>(combbf, pwt, pb, x, yb);
    rmsnorm_kernel<<<T_TOK, 256, 0, stream>>>(yb, nw, out);
}
